// PySelfMultiheadAttn_16887811408047
// MI455X (gfx1250) — matrix operand order repaired and checked by reading, not compiled, not run on hardware
//
#include <hip/hip_runtime.h>
#include <hip/hip_bf16.h>

// ---------------------------------------------------------------------------
// Self multi-head attention, T=2048 B=4 E=1024 H=16 HD=64, fp32 in/out.
// bf16 operands + f32 WMMA accumulation (v_wmma_f32_16x16x32_bf16),
// flash-attention with online softmax, TDM (tensor_load_to_lds) K-tile
// staging, ds_swizzle xor-reductions, LDS-staged V tiles.
// ---------------------------------------------------------------------------

typedef __bf16 bf16;
typedef __attribute__((ext_vector_type(16))) __bf16 v16bf;
typedef __attribute__((ext_vector_type(8)))  __bf16 v8bf;
typedef __attribute__((ext_vector_type(4)))  __bf16 v4bf;
typedef __attribute__((ext_vector_type(8)))  float  v8f;
typedef __attribute__((ext_vector_type(4)))  unsigned int v4u;
typedef __attribute__((ext_vector_type(8)))  int    v8i;
typedef __attribute__((ext_vector_type(4)))  int    v4i;

#define T_DIM   2048
#define B_DIM   4
#define E_DIM   1024
#define H_DIM   16
#define HD_DIM  64
#define NROWS   (T_DIM * B_DIM)   // 8192 token rows
#define F_DIM   (3 * E_DIM)       // 3072 qkv features
#define ATTN_SCALE 0.125f         // 1/sqrt(64)

#if __has_builtin(__builtin_amdgcn_tensor_load_to_lds)
#define HAVE_TDM 1
#else
#define HAVE_TDM 0
#endif

__device__ __forceinline__ v8f wmma_bf16(v16bf a, v16bf b, v8f c) {
  // D = A(16x32 bf16) * B(32x16 bf16) + C(16x16 f32)
  return __builtin_amdgcn_wmma_f32_16x16x32_bf16(
      /*neg_a=*/false, a, /*neg_b=*/false, b,
      /*c_mod=*/(short)0, c, /*reuse_a=*/false, /*reuse_b=*/false);
}

__device__ __forceinline__ v16bf cat16(v8bf lo, v8bf hi) {
  return __builtin_shufflevector(lo, hi, 0,1,2,3,4,5,6,7,8,9,10,11,12,13,14,15);
}

// ds_swizzle_b32 xor-lane exchange (group-of-32 mode: offset = xor<<10 | 0x1F)
template <int PATT>
__device__ __forceinline__ float swz(float x) {
  return __builtin_bit_cast(float,
      __builtin_amdgcn_ds_swizzle(__builtin_bit_cast(int, x), PATT));
}
__device__ __forceinline__ float rowmax16(float m) {  // max across 16-lane groups
  m = fmaxf(m, swz<0x041F>(m));   // SWAPX1
  m = fmaxf(m, swz<0x081F>(m));   // SWAPX2
  m = fmaxf(m, swz<0x101F>(m));   // SWAPX4
  m = fmaxf(m, swz<0x201F>(m));   // SWAPX8
  return m;
}
__device__ __forceinline__ float rowsum16(float s) {
  s += swz<0x041F>(s);
  s += swz<0x081F>(s);
  s += swz<0x101F>(s);
  s += swz<0x201F>(s);
  return s;
}

// A fragment, 16x32 bf16, row-major source with leading dim ldk.
// lane<16 -> row=lane, K 0..15; lane>=16 -> row=lane-16, K 16..31.
__device__ __forceinline__ v16bf load_a_frag(const bf16* __restrict__ base, int ldk) {
  const int lane = threadIdx.x & 31;
  const bf16* p = base + (size_t)(lane & 15) * ldk + (lane >> 4) * 16;
  v8bf lo = *(const v8bf*)(p);
  v8bf hi = *(const v8bf*)(p + 8);
  return cat16(lo, hi);
}

// B fragment, 32x16 bf16, from a row-major (Ncols x K) "weight" matrix.
// lane<16 -> col=lane, K 0..15; lane>=16 -> col=lane-16, K 16..31.
__device__ __forceinline__ v16bf load_b_frag(const bf16* __restrict__ base, int ldk) {
  const int lane = threadIdx.x & 31;
  return *(const v16bf*)(base + (size_t)(lane & 15) * ldk + (lane >> 4) * 16);
}

// ---------------------------------------------------------------------------
// fp32 -> bf16 cast (n divisible by 4)
// ---------------------------------------------------------------------------
__global__ __launch_bounds__(256) void cast_f32_bf16(const float* __restrict__ in,
                                                     bf16* __restrict__ out, int n) {
  int i = (blockIdx.x * 256 + threadIdx.x) * 4;
  if (i < n) {
    float4 v = *(const float4*)(in + i);
    v4bf o;
    o[0] = (bf16)v.x; o[1] = (bf16)v.y; o[2] = (bf16)v.z; o[3] = (bf16)v.w;
    *(v4bf*)(out + i) = o;
  }
}

// ---------------------------------------------------------------------------
// C(MxN,f32 acc) = A(MxK) * W(NxK)^T, bf16 operands, WMMA 16x16x32.
// Block: 256 thr = 8 waves -> tile 128x64; wave tile 32x32 (2x2 WMMA tiles).
// ---------------------------------------------------------------------------
template <bool OUT_BF16>
__global__ __launch_bounds__(256) void gemm_xwt(const bf16* __restrict__ A,
                                                const bf16* __restrict__ W,
                                                void* __restrict__ Cout,
                                                int M, int N, int K) {
  const int w    = threadIdx.x >> 5;
  const int lane = threadIdx.x & 31;
  const int rowBase = blockIdx.y * 128 + (w >> 1) * 32;
  const int colBase = blockIdx.x * 64  + (w & 1) * 32;

  v8f acc[2][2] = {};

  for (int k0 = 0; k0 < K; k0 += 32) {
    if (k0 + 32 < K) {  // global_prefetch_b8 for next K slab
      __builtin_prefetch(A + (size_t)rowBase * K + k0 + 32, 0, 1);
      __builtin_prefetch(W + (size_t)colBase * K + k0 + 32, 0, 1);
    }
    v16bf af[2], bf_[2];
#pragma unroll
    for (int mi = 0; mi < 2; ++mi)
      af[mi] = load_a_frag(A + (size_t)(rowBase + mi * 16) * K + k0, K);
#pragma unroll
    for (int ni = 0; ni < 2; ++ni)
      bf_[ni] = load_b_frag(W + (size_t)(colBase + ni * 16) * K + k0, K);
#pragma unroll
    for (int mi = 0; mi < 2; ++mi)
#pragma unroll
      for (int ni = 0; ni < 2; ++ni)
        acc[mi][ni] = wmma_bf16(af[mi], bf_[ni], acc[mi][ni]);
  }

  // C layout: VGPR r -> row r (lanes 0..15) / row r+8 (lanes 16..31), col=lane&15
  const int colL  = lane & 15;
  const int rhalf = (lane >> 4) * 8;
#pragma unroll
  for (int mi = 0; mi < 2; ++mi)
#pragma unroll
    for (int ni = 0; ni < 2; ++ni)
#pragma unroll
      for (int r = 0; r < 8; ++r) {
        size_t idx = (size_t)(rowBase + mi * 16 + r + rhalf) * N + colBase + ni * 16 + colL;
        if (OUT_BF16) ((bf16*)Cout)[idx]  = (bf16)acc[mi][ni][r];
        else          ((float*)Cout)[idx] = acc[mi][ni][r];
      }
}

// ---------------------------------------------------------------------------
// Flash attention: one head per blockIdx.y, 64 query rows per block,
// 4 waves x 16 rows. Key tiles of 32. K tile DMA'd into LDS by the Tensor
// Data Mover (descriptor-driven, padded to a 72-element row stride); V tile
// staged transposed by VALU so PV B-fragments are contiguous ds_load_b128.
// QKV row layout: row r = t*4+b, q at col h*192+d, k at +64, v at +128.
// ---------------------------------------------------------------------------
__global__ __launch_bounds__(128) void attn_kernel(const bf16* __restrict__ QKV,
                                                   bf16* __restrict__ CTX) {
  constexpr int LDK = 72;  // bf16 elems per sK row (64 + 8 pad = 36 dwords)
  constexpr int LDV = 40;  // bf16 elems per sVt row (32 keys + pad)
  constexpr int LDP = 40;  // bf16 elems per sP row (32 + pad)
  __shared__ __attribute__((aligned(128))) bf16 sK[32 * LDK];  // 32 keys x 64 d
  __shared__ __attribute__((aligned(128))) bf16 sVt[64 * LDV]; // 64 d x 32 keys
  __shared__ __attribute__((aligned(128))) bf16 sP[4][16 * LDP];

  const int tid  = threadIdx.x;
  const int w    = tid >> 5;
  const int lane = tid & 31;
  const int rl    = lane & 15;
  const int khalf = lane >> 4;

  const int n = blockIdx.y;          // head-row index in [0,64)
  const int b = n >> 4, h = n & 15;
  const int q0 = blockIdx.x * 64 + w * 16;

  const bf16* Qb = QKV + h * 192;
  const bf16* Kb = QKV + h * 192 + 64;
  const bf16* Vb = QKV + h * 192 + 128;

  // Q fragments for this wave's 16 rows, d split into 2 K=32 slabs.
  v16bf qf[2];
  {
    const bf16* qp = Qb + (size_t)((q0 + rl) * 4 + b) * F_DIM + khalf * 16;
#pragma unroll
    for (int f = 0; f < 2; ++f) {
      v8bf lo = *(const v8bf*)(qp + f * 32);
      v8bf hi = *(const v8bf*)(qp + f * 32 + 8);
      qf[f] = cat16(lo, hi);
    }
  }

#if HAVE_TDM
  // Tensor DMA descriptor pieces that do not change across iterations.
  const unsigned lds_off = (unsigned)(uintptr_t)&sK[0];  // flat addr low 32b = LDS offset
  // Group1: wg_mask=0 | data_size=2B | pad_enable | pad_interval=32dw | pad_amount=4dw
  const int g1d0 = (1 << 16) | (1 << 20) | (4 << 22) | (3 << 25);
  const int g1d1 = (64 << 16);          // tensor_dim0 = 64 (low 16 bits, at bits 63:48)
  const int g1d3 = 0x0010 | (64 << 16); // tensor_dim1 hi16 (=1<<20) | tile_dim0 = 64
  const int g1d4 = 32;                  // tile_dim1 = 32 keys, tile_dim2 = 0
  const int g1d5 = 4 * F_DIM;           // tensor_dim0_stride = 12288 elements
#endif

  v8f acc[4] = {};                   // 16 rows x 64 d context accumulator
  float rowmax[8], rowsum[8];
#pragma unroll
  for (int r = 0; r < 8; ++r) { rowmax[r] = -__builtin_inff(); rowsum[r] = 0.0f; }

  for (int s0 = 0; s0 < T_DIM; s0 += 32) {
    __syncthreads();  // previous tile fully consumed before overwrite

#if HAVE_TDM
    if (w == 0) {  // one TDM issue per key tile (EXEC-gated; TDM ignores EXEC)
      unsigned long long ga =
          (unsigned long long)(uintptr_t)(Kb + (size_t)(s0 * 4 + b) * F_DIM);
      v4u g0 = { 1u /*count=1*/, lds_off, (unsigned)ga,
                 (unsigned)((ga >> 32) & 0x01FFFFFFu) | 0x80000000u /*type=2*/ };
      v8i g1 = { g1d0, g1d1, 0, g1d3, g1d4, g1d5, 0, 0 };
      v4i g2 = { 0, 0, 0, 0 };
      v4i g3 = { 0, 0, 0, 0 };
#if defined(__clang_major__) && __clang_major__ >= 23
      v8i g4 = { 0, 0, 0, 0, 0, 0, 0, 0 };
      __builtin_amdgcn_tensor_load_to_lds(g0, g1, g2, g3, g4, 0);
#else
      __builtin_amdgcn_tensor_load_to_lds(g0, g1, g2, g3, 0);
#endif
    }
#else
    // Fallback: cooperative K tile load, 32 keys x 64 d
    for (int i = tid; i < 32 * 8; i += 128) {
      int key = i >> 3, dc = (i & 7) * 8;
      v8bf kv = *(const v8bf*)(Kb + (size_t)((s0 + key) * 4 + b) * F_DIM + dc);
      *(v8bf*)(&sK[key * LDK + dc]) = kv;
    }
#endif

    // Cooperative V tile, transposed into sVt[d][key]
    for (int i = tid; i < 32 * 8; i += 128) {
      int key = i >> 3, dc = (i & 7) * 8;
      v8bf vv = *(const v8bf*)(Vb + (size_t)((s0 + key) * 4 + b) * F_DIM + dc);
#pragma unroll
      for (int j = 0; j < 8; ++j) sVt[(dc + j) * LDV + key] = vv[j];
    }

#if HAVE_TDM
    __builtin_amdgcn_s_wait_tensorcnt(0);  // no-op for waves that issued none
#endif
    __syncthreads();

    // S = Q * K^T : Sa = keys s0..s0+15, Sb = keys s0+16..s0+31
    v8f Sa = {}, Sb = {};
#pragma unroll
    for (int f = 0; f < 2; ++f) {
      v16bf ka = *(const v16bf*)(&sK[rl * LDK + f * 32 + khalf * 16]);
      v16bf kb = *(const v16bf*)(&sK[(16 + rl) * LDK + f * 32 + khalf * 16]);
      Sa = wmma_bf16(qf[f], ka, Sa);
      Sb = wmma_bf16(qf[f], kb, Sb);
    }

    // Online softmax. Row m of a C tile lives in VGPR m across one 16-lane
    // group -> ds_swizzle xor-reductions with masks 1,2,4,8.
    float mt[8];
#pragma unroll
    for (int r = 0; r < 8; ++r) {
      Sa[r] *= ATTN_SCALE;
      Sb[r] *= ATTN_SCALE;
      mt[r] = rowmax16(fmaxf(Sa[r], Sb[r]));
    }
#pragma unroll
    for (int r = 0; r < 8; ++r) {
      float nm   = fmaxf(rowmax[r], mt[r]);
      float corr = __expf(rowmax[r] - nm);
      rowmax[r]  = nm;
      float pa = __expf(Sa[r] - nm);
      float pb = __expf(Sb[r] - nm);
      Sa[r] = pa; Sb[r] = pb;
      rowsum[r] = rowsum[r] * corr + rowsum16(pa + pb);
#pragma unroll
      for (int dt = 0; dt < 4; ++dt) acc[dt][r] *= corr;
    }

    // Re-layout P: C-tile -> bf16 A-fragment via per-wave LDS
    bf16* pT = sP[w];
#pragma unroll
    for (int r = 0; r < 8; ++r) {
      int prow = r + ((lane >> 4) << 3);
      pT[prow * LDP + rl]      = (bf16)Sa[r];
      pT[prow * LDP + rl + 16] = (bf16)Sb[r];
    }
    v8bf plo = *(const v8bf*)(&pT[rl * LDP + khalf * 16]);
    v8bf phi = *(const v8bf*)(&pT[rl * LDP + khalf * 16 + 8]);
    v16bf pf = cat16(plo, phi);

    // ctx += P(16x32) * V(32x64), V B-fragments contiguous from sVt
#pragma unroll
    for (int dt = 0; dt < 4; ++dt) {
      v16bf vf = *(const v16bf*)(&sVt[(dt * 16 + rl) * LDV + khalf * 16]);
      acc[dt] = wmma_bf16(pf, vf, acc[dt]);
    }
  }

  // Epilogue: ctx[t, b, h*64+d] in bf16 for the output projection
#pragma unroll
  for (int r = 0; r < 8; ++r) {
    float inv = 1.0f / rowsum[r];
    int   t   = q0 + r + ((lane >> 4) << 3);
    size_t off = (size_t)(t * 4 + b) * E_DIM + h * 64;
#pragma unroll
    for (int dt = 0; dt < 4; ++dt)
      CTX[off + dt * 16 + rl] = (bf16)(acc[dt][r] * inv);
  }
}

// ---------------------------------------------------------------------------
// Launch
// ---------------------------------------------------------------------------
extern "C" void kernel_launch(void* const* d_in, const int* in_sizes, int n_in,
                              void* d_out, int out_size, void* d_ws, size_t ws_size,
                              hipStream_t stream) {
  const float* X  = (const float*)d_in[0];   // (T,B,E)  8192x1024
  const float* Wq = (const float*)d_in[1];   // (3E,E)   3072x1024
  const float* Wo = (const float*)d_in[2];   // (E,E)    1024x1024
  float* out = (float*)d_out;                // (T,B,E)  fp32

  char* ws = (char*)d_ws;
  const size_t MB = 1024 * 1024;
  bf16* Xbf   = (bf16*)(ws + 0);          // 16 MiB
  bf16* Wqbf  = (bf16*)(ws + 16 * MB);    //  6 MiB
  bf16* Wobf  = (bf16*)(ws + 22 * MB);    //  2 MiB
  bf16* QKVbf = (bf16*)(ws + 24 * MB);    // 48 MiB
  bf16* CTXbf = (bf16*)(ws + 72 * MB);    // 16 MiB   (total 88 MiB)

  const int nX = NROWS * E_DIM, nWq = F_DIM * E_DIM, nWo = E_DIM * E_DIM;
  cast_f32_bf16<<<nX  / 1024, 256, 0, stream>>>(X,  Xbf,  nX);
  cast_f32_bf16<<<nWq / 1024, 256, 0, stream>>>(Wq, Wqbf, nWq);
  cast_f32_bf16<<<nWo / 1024, 256, 0, stream>>>(Wo, Wobf, nWo);

  // QKV = X * Wqkv^T   (8192 x 3072)
  gemm_xwt<true><<<dim3(F_DIM / 64, NROWS / 128), 256, 0, stream>>>(
      Xbf, Wqbf, QKVbf, NROWS, F_DIM, E_DIM);

  // Flash attention: grid = (q blocks of 64, 64 heads)
  attn_kernel<<<dim3(T_DIM / 64, B_DIM * H_DIM), 128, 0, stream>>>(QKVbf, CTXbf);

  // out = ctx * Wout^T  (8192 x 1024), fp32 result
  gemm_xwt<false><<<dim3(E_DIM / 64, NROWS / 128), 256, 0, stream>>>(
      CTXbf, Wobf, out, NROWS, E_DIM, E_DIM);
}
